// PointFlowModuleWithCornerEdgeSampling_76338748720085
// MI455X (gfx1250) — matrix operand order repaired and checked by reading, not compiled, not run on hardware
//
#include <hip/hip_runtime.h>

typedef __attribute__((ext_vector_type(16))) __bf16 v16bf;
typedef __attribute__((ext_vector_type(8)))  float  v8f;
typedef __attribute__((ext_vector_type(4)))  unsigned int u32x4;
typedef __attribute__((ext_vector_type(8)))  int i32x8;
typedef __attribute__((ext_vector_type(4)))  int i32x4;

#define NB   16
#define CIN  256
#define COUT 512
#define HH   64
#define WH   64
#define HL   128
#define WL   128
#define NPTS 128

// ---------------------------------------------------------------------------
// K0: pack conv1 weights (both branches) to bf16, layout [khw][co(512)][ci(256)]
// ---------------------------------------------------------------------------
__global__ void __launch_bounds__(256) pack_w_kernel(const float* __restrict__ we1,
                                                     const float* __restrict__ wc1,
                                                     __bf16* __restrict__ wpk) {
    int i = blockIdx.x * 256 + threadIdx.x;          // total 9*512*256
    if (i >= 9 * 512 * 256) return;
    int ci  = i & 255;
    int co  = (i >> 8) & 511;
    int khw = i >> 17;
    float v = (co < 256) ? we1[((size_t)co * 256 + ci) * 9 + khw]
                         : wc1[((size_t)(co - 256) * 256 + ci) * 9 + khw];
    wpk[i] = (__bf16)v;
}

// ---------------------------------------------------------------------------
// K0b: fold BN into per-channel scale/bias (co<256 edge, co>=256 corner)
// ---------------------------------------------------------------------------
__global__ void __launch_bounds__(256) bn_prep_kernel(
    const float* ge, const float* be, const float* me, const float* ve,
    const float* gc, const float* bc, const float* mc, const float* vc,
    float* __restrict__ scale, float* __restrict__ bias) {
    int co = blockIdx.x * 256 + threadIdx.x;
    if (co >= 512) return;
    float g, b, m, v;
    if (co < 256) { g = ge[co]; b = be[co]; m = me[co]; v = ve[co]; }
    else { int c = co - 256; g = gc[c]; b = bc[c]; m = mc[c]; v = vc[c]; }
    float inv = g * rsqrtf(v + 1e-5f);
    scale[co] = inv;
    bias[co]  = b - m * inv;
}

// ---------------------------------------------------------------------------
// K1: fp32 -> bf16 convert (x_high)
// ---------------------------------------------------------------------------
__global__ void __launch_bounds__(256) cvt_bf16_kernel(const float* __restrict__ x,
                                                       __bf16* __restrict__ o, int n) {
    int i = blockIdx.x * 256 + threadIdx.x;
    if (i < n) o[i] = (__bf16)x[i];
}

// ---------------------------------------------------------------------------
// K2: fused conv3x3(pad1) for both branches as implicit GEMM with WMMA bf16.
//     M = co(512), N = pos(4096/image), K = 9*256.  Tile: WG = 128co x 128pos,
//     8 waves (2M x 4N), wave = 64co x 32pos = 4x2 accumulators.
//     A-tile (weights) staged by Tensor Data Mover, double-buffered, with TDM
//     LDS padding reproducing the stride-34 (17-dword) conflict-free rows.
//     B-tile (im2col window, needs boundary zeros) staged manually.
//     Epilogue: BN + ReLU, store bf16 Y1[n][co][h][w].
// ---------------------------------------------------------------------------
__global__ void __launch_bounds__(256) conv1_wmma_kernel(
    const __bf16* __restrict__ xh, const __bf16* __restrict__ wpk,
    const float* __restrict__ bnsc, const float* __restrict__ bnbi,
    __bf16* __restrict__ y1) {
    const int n    = blockIdx.z;
    const int coB  = blockIdx.y * 128;
    const int posB = blockIdx.x * 128;          // covers 2 output rows
    const int h0   = posB >> 6;

    const int t    = threadIdx.x;
    const int lane = t & 31;
    const int wid  = t >> 5;
    const int wm   = wid >> 2;                  // 0..1 over M
    const int wn   = wid & 3;                   // 0..3 over N

    __shared__ alignas(16) __bf16 Asm[2][128 * 34];   // double-buffered TDM dest
    __shared__ alignas(16) __bf16 Bsm[128 * 34];

    // ---- TDM descriptor (wave-uniform) -----------------------------------
    const unsigned lds0 = (unsigned)(size_t)(&Asm[0][0]);
    const unsigned lds1 = (unsigned)(size_t)(&Asm[1][0]);
    const unsigned long long wbase = (unsigned long long)(size_t)wpk;

    i32x8 g1;
    // data_size=2B(1<<16), pad_enable(1<<20), pad_interval=16dw(3<<22), pad=1dw(0<<25)
    g1[0] = (1 << 16) | (1 << 20) | (3 << 22);
    g1[1] = (int)((256u & 0xFFFFu) << 16);      // tensor_dim0 = 256 (lo16)
    g1[2] = (int)((4608u & 0xFFFFu) << 16);     // dim0 hi16=0 | tensor_dim1=4608 lo16
    g1[3] = (int)(32u << 16);                   // dim1 hi16=0 | tile_dim0 = 32
    g1[4] = 128;                                // tile_dim1 = 128 | tile_dim2 = 0
    g1[5] = 256;                                // tensor_dim0_stride = 256 (lo32)
    g1[6] = 0;
    g1[7] = 0;
    i32x4 gz4;
    gz4[0] = 0; gz4[1] = 0; gz4[2] = 0; gz4[3] = 0;
    i32x8 gz8;
    gz8[0] = 0; gz8[1] = 0; gz8[2] = 0; gz8[3] = 0;
    gz8[4] = 0; gz8[5] = 0; gz8[6] = 0; gz8[7] = 0;

    auto issueA = [&](int ks, int buf) {
        int khw = ks >> 3, ci0 = (ks & 7) * 32;
        unsigned long long ga =
            wbase + (((unsigned long long)khw * 512 + coB) * 256 + ci0) * 2ull;
        u32x4 g0;
        g0[0] = 1u;                                             // count=1, user desc
        g0[1] = buf ? lds1 : lds0;                              // lds_addr
        g0[2] = (unsigned)ga;                                   // global_addr lo
        g0[3] = (unsigned)((ga >> 32) & 0x01FFFFFFu) | (2u << 30); // hi | type=2
        __builtin_amdgcn_tensor_load_to_lds(g0, g1, gz4, gz4, gz8, 0);
    };

    v8f acc[4][2];
#pragma unroll
    for (int a = 0; a < 4; a++)
#pragma unroll
        for (int b = 0; b < 2; b++)
#pragma unroll
            for (int q = 0; q < 8; q++) acc[a][b][q] = 0.f;

    const int mlo  = lane & 15;
    const int hiA8 = (lane >> 4) * 8;
    const int hiB16 = (lane >> 4) * 16;

    if (wid == 0) issueA(0, 0);                 // prologue: prefetch first A tile

    for (int ks = 0; ks < 72; ++ks) {
        const int cur = ks & 1, nxt = cur ^ 1;
        const int khw = ks >> 3;                // 0..8 filter tap
        const int ci0 = (ks & 7) * 32;          // ci chunk
        const int kh  = khw / 3, kw = khw % 3;

        __syncthreads();                        // prev-iter LDS reads complete
        if (wid == 0 && ks + 1 < 72) issueA(ks + 1, nxt);   // async next A tile

        // ---- stage B: shifted input tile  Bsm[pos][ci'] ----
        {
            int ci   = t >> 3;                  // 0..31
            int pos0 = (t & 7) * 16;            // stays within one row of 64
            int r    = pos0 >> 6;
            int y    = h0 + r + kh - 1;
            bool yok = (unsigned)y < 64u;
            const __bf16* src = xh + (((size_t)n * 256 + ci0 + ci) * 64 + (yok ? y : 0)) * 64;
#pragma unroll
            for (int i = 0; i < 16; i++) {
                int w = (pos0 + i) & 63;
                int x = w + kw - 1;
                __bf16 v = (__bf16)0.f;
                if (yok && (unsigned)x < 64u) v = src[x];
                Bsm[(pos0 + i) * 34 + ci] = v;
            }
        }
        if (wid == 0) {                         // current A tile landed?
            if (ks + 1 < 72) __builtin_amdgcn_s_wait_tensorcnt(1);
            else             __builtin_amdgcn_s_wait_tensorcnt(0);
        }
        __syncthreads();                        // B staged + A(cur) DMA complete

        // ---- fragments (contiguous k = 16*(lane/16) + s, A and B alike) ----
        const __bf16* Acur = &Asm[cur][0];
        v16bf afr[4], bfr[2];
#pragma unroll
        for (int fi = 0; fi < 4; fi++) {
            int m = wm * 64 + fi * 16 + mlo;
            const unsigned* row = (const unsigned*)(Acur + m * 34);
            union { v16bf v; unsigned u[8]; } U;
#pragma unroll
            for (int v8 = 0; v8 < 8; v8++) {
                int k = hiB16 + 2 * v8;
                U.u[v8] = row[k >> 1];
            }
            afr[fi] = U.v;
        }
#pragma unroll
        for (int fj = 0; fj < 2; fj++) {
            int p = wn * 32 + fj * 16 + mlo;
            const unsigned* row = (const unsigned*)(Bsm + p * 34);
            union { v16bf v; unsigned u[8]; } U;
#pragma unroll
            for (int v8 = 0; v8 < 8; v8++) U.u[v8] = row[(hiB16 >> 1) + v8];
            bfr[fj] = U.v;
        }
#pragma unroll
        for (int fi = 0; fi < 4; fi++)
#pragma unroll
            for (int fj = 0; fj < 2; fj++)
                acc[fi][fj] = __builtin_amdgcn_wmma_f32_16x16x32_bf16(
                    false, afr[fi], false, bfr[fj], (short)0, acc[fi][fj], false, false);
    }

    // ---- epilogue: BN + ReLU, store bf16 ----
    const int mhi = (lane >> 4) * 8;
#pragma unroll
    for (int fi = 0; fi < 4; fi++) {
#pragma unroll
        for (int fj = 0; fj < 2; fj++) {
            int p = posB + wn * 32 + fj * 16 + mlo;
            int h = p >> 6, w = p & 63;
#pragma unroll
            for (int v8 = 0; v8 < 8; v8++) {
                int co = coB + wm * 64 + fi * 16 + v8 + mhi;
                float val = fmaf(acc[fi][fj][v8], bnsc[co], bnbi[co]);
                val = val > 0.f ? val : 0.f;
                y1[(((size_t)n * COUT + co) * 64 + h) * 64 + w] = (__bf16)val;
            }
        }
    }
}

// ---------------------------------------------------------------------------
// K3: edge_pred = conv3x3(Y1[:,0:256], w_e2)  (one thread per output pixel)
// ---------------------------------------------------------------------------
__global__ void __launch_bounds__(256) edge_pred_kernel(const __bf16* __restrict__ y1,
                                                        const float* __restrict__ we2,
                                                        float* __restrict__ out) {
    __shared__ float wsm[2304];
    int t = threadIdx.x;
    for (int j = t; j < 2304; j += 256) wsm[j] = we2[j];
    __syncthreads();
    int pix = blockIdx.x * 256 + t;             // 65536 total
    int n = pix >> 12, hw = pix & 4095;
    int h = hw >> 6, w = hw & 63;
    const __bf16* yb = y1 + (size_t)n * COUT * 4096;  // edge half: channels 0..255
    float s = 0.f;
    for (int c = 0; c < 256; c++) {
        const __bf16* row = yb + (size_t)c * 4096;
        const float* wc = wsm + c * 9;
#pragma unroll
        for (int kh = 0; kh < 3; kh++) {
            int y = h + kh - 1;
            if ((unsigned)y >= 64u) continue;
#pragma unroll
            for (int kw = 0; kw < 3; kw++) {
                int x = w + kw - 1;
                if ((unsigned)x >= 64u) continue;
                s = fmaf((float)row[y * 64 + x], wc[kh * 3 + kw], s);
            }
        }
    }
    out[pix] = s;
}

// ---------------------------------------------------------------------------
// K4: corner_pred = conv1x1(Y1[:,256:512], w_c2)
// ---------------------------------------------------------------------------
__global__ void __launch_bounds__(256) corner_pred_kernel(const __bf16* __restrict__ y1,
                                                          const float* __restrict__ wc2,
                                                          float* __restrict__ out) {
    __shared__ float wsm[256];
    int t = threadIdx.x;
    if (t < 256) wsm[t] = wc2[t];
    __syncthreads();
    int pix = blockIdx.x * 256 + t;
    int n = pix >> 12, hw = pix & 4095;
    const __bf16* yb = y1 + ((size_t)n * COUT + 256) * 4096 + hw;
    float s = 0.f;
    for (int c = 0; c < 256; c++) s = fmaf((float)yb[(size_t)c * 4096], wsm[c], s);
    out[pix] = s;
}

// ---------------------------------------------------------------------------
// K5: top-128 of 4096 per (n, branch); tie-break = lowest index (lax.top_k)
// ---------------------------------------------------------------------------
__global__ void __launch_bounds__(256) topk_kernel(const float* __restrict__ preds,
                                                   int* __restrict__ topk) {
    int n = blockIdx.x, br = blockIdx.y;
    const float* pred = preds + (size_t)br * (NB * 4096) + (size_t)n * 4096;
    __shared__ float vals[4096];
    __shared__ float rv[256];
    __shared__ int   ri[256];
    int t = threadIdx.x;
    for (int j = t; j < 4096; j += 256) vals[j] = pred[j];
    __syncthreads();
    int* out = topk + (br * NB + n) * NPTS;
    for (int k = 0; k < NPTS; k++) {
        float best = -__builtin_inff();
        int bi = 0;
        for (int j = t; j < 4096; j += 256) {
            float v = vals[j];
            if (v > best) { best = v; bi = j; }     // first (lowest) index kept on ties
        }
        rv[t] = best; ri[t] = bi;
        __syncthreads();
        for (int off = 128; off > 0; off >>= 1) {
            if (t < off) {
                float vo = rv[t + off]; int io = ri[t + off];
                if (vo > rv[t] || (vo == rv[t] && io < ri[t])) { rv[t] = vo; ri[t] = io; }
            }
            __syncthreads();
        }
        if (t == 0) { out[k] = ri[0]; vals[ri[0]] = -__builtin_inff(); }
        __syncthreads();
    }
}

// ---------------------------------------------------------------------------
// K6a: gather hf (exact grid hit on x_high) and lf (exact 2x2 quarter-weight
//      average on x_low), emit bf16 + fp32 lf and low-grid scatter indices.
// ---------------------------------------------------------------------------
__global__ void __launch_bounds__(256) gather_kernel(
    const float* __restrict__ xhigh, const float* __restrict__ xlow,
    const int* __restrict__ topk, __bf16* __restrict__ hfb,
    __bf16* __restrict__ lfb, float* __restrict__ lf32,
    int* __restrict__ lowidx) {
    int n = blockIdx.x, br = blockIdx.y;
    __shared__ int sxs[NPTS], sys[NPTS];
    int t = threadIdx.x;
    const int* idx = topk + (br * NB + n) * NPTS;
    if (t < NPTS) {
        int id = idx[t];
        int xs = id & 63, ys = id >> 6;
        sxs[t] = xs; sys[t] = ys;
        lowidx[(br * NB + n) * NPTS + t] = ys * 256 + xs * 2;   // (2*ys)*128 + 2*xs
    }
    __syncthreads();
    size_t base = ((size_t)br * NB + n) * CIN * NPTS;
    const float* xh = xhigh + (size_t)n * CIN * HH * WH;
    const float* xl = xlow + (size_t)n * CIN * HL * WL;
    for (int e = t; e < CIN * NPTS; e += 256) {
        int c = e >> 7, p = e & 127;
        int xs = sxs[p], ys = sys[p];
        float h = xh[(c * 64 + ys) * 64 + xs];
        const float* blk = xl + ((size_t)c * HL + 2 * ys) * WL + 2 * xs;
        float l = 0.25f * (blk[0] + blk[1] + blk[WL] + blk[WL + 1]);
        hfb[base + e]  = (__bf16)h;
        lfb[base + e]  = (__bf16)l;
        lf32[base + e] = l;
    }
}

// ---------------------------------------------------------------------------
// K6b: attention per (n,branch) with WMMA bf16.
//  Phase1: S[q][p] = sum_c lf[c][q]*hf[c][p]   (128x128, K=256)
//  softmax over p -> P (bf16, stride 132)
//  Phase2: F[c][i] = sum_j hf[c][j]*P[i][j]    (256x128, K=128); += lf32
// ---------------------------------------------------------------------------
__global__ void __launch_bounds__(256) attn_wmma_kernel(
    const __bf16* __restrict__ hfb, const __bf16* __restrict__ lfb,
    const float* __restrict__ lf32, float* __restrict__ fused) {
    int n = blockIdx.x, br = blockIdx.y;
    size_t base = ((size_t)br * NB + n) * CIN * NPTS;
    const __bf16* hf = hfb + base;
    const __bf16* lf = lfb + base;
    const float* lfp = lf32 + base;
    float* fup = fused + base;

    __shared__ alignas(16) float  Sbuf[128 * 128];     // 64KB, aliased by GEMM tiles
    __shared__ alignas(16) __bf16 Pbuf[128 * 132];     // softmax probs, padded rows

    __bf16* Atile = (__bf16*)Sbuf;          // phase1 A: 128x34
    __bf16* Btile = Atile + 128 * 34;       // phase1 B: 128x34
    __bf16* A2tile = (__bf16*)Sbuf;         // phase2 A: 256x34 (S is dead by then)

    const int t    = threadIdx.x;
    const int lane = t & 31;
    const int wid  = t >> 5;
    const int mlo  = lane & 15;
    const int mhi  = (lane >> 4) * 8;
    const int hiA8 = (lane >> 4) * 8;
    const int hiB16 = (lane >> 4) * 16;

    // ---------------- phase 1: S = lf^T * hf ----------------
    {
        const int wm = wid >> 2;            // 0..1 over q
        const int wn = wid & 3;             // 0..3 over p
        v8f acc[4][2];
#pragma unroll
        for (int a = 0; a < 4; a++)
#pragma unroll
            for (int b = 0; b < 2; b++)
#pragma unroll
                for (int q = 0; q < 8; q++) acc[a][b][q] = 0.f;

        for (int cs = 0; cs < 8; cs++) {
            int c0 = cs * 32;
            __syncthreads();
            {   // Atile[q][c'] = lf[c0+c'][q]  (transposed gather)
                int q = t >> 1, cst = (t & 1) * 16;
#pragma unroll
                for (int i = 0; i < 16; i++)
                    Atile[q * 34 + cst + i] = lf[(size_t)(c0 + cst + i) * NPTS + q];
            }
            {   // Btile[p][c'] = hf[c0+c'][p]
                int p = t >> 1, cst = (t & 1) * 16;
#pragma unroll
                for (int i = 0; i < 16; i++)
                    Btile[p * 34 + cst + i] = hf[(size_t)(c0 + cst + i) * NPTS + p];
            }
            __syncthreads();

            v16bf afr[4], bfr[2];
#pragma unroll
            for (int fi = 0; fi < 4; fi++) {
                int m = wm * 64 + fi * 16 + mlo;
                const unsigned* row = (const unsigned*)(Atile + m * 34);
                union { v16bf v; unsigned u[8]; } U;
#pragma unroll
                for (int v8 = 0; v8 < 8; v8++) {
                    int k = hiB16 + 2 * v8;
                    U.u[v8] = row[k >> 1];
                }
                afr[fi] = U.v;
            }
#pragma unroll
            for (int fj = 0; fj < 2; fj++) {
                int p = wn * 32 + fj * 16 + mlo;
                const unsigned* row = (const unsigned*)(Btile + p * 34);
                union { v16bf v; unsigned u[8]; } U;
#pragma unroll
                for (int v8 = 0; v8 < 8; v8++) U.u[v8] = row[(hiB16 >> 1) + v8];
                bfr[fj] = U.v;
            }
#pragma unroll
            for (int fi = 0; fi < 4; fi++)
#pragma unroll
                for (int fj = 0; fj < 2; fj++)
                    acc[fi][fj] = __builtin_amdgcn_wmma_f32_16x16x32_bf16(
                        false, afr[fi], false, bfr[fj], (short)0, acc[fi][fj], false, false);
        }

        __syncthreads();                    // tiles dead, write S into Sbuf
#pragma unroll
        for (int fi = 0; fi < 4; fi++)
#pragma unroll
            for (int fj = 0; fj < 2; fj++) {
                int p = wn * 32 + fj * 16 + mlo;
#pragma unroll
                for (int v8 = 0; v8 < 8; v8++) {
                    int q = wm * 64 + fi * 16 + v8 + mhi;
                    Sbuf[q * 128 + p] = acc[fi][fj][v8];
                }
            }
        __syncthreads();
    }

    // ---------------- softmax over p, emit bf16 P ----------------
    if (t < 128) {
        float* row = Sbuf + t * 128;
        float m = row[0];
        for (int j = 1; j < 128; j++) m = fmaxf(m, row[j]);
        float s = 0.f;
        for (int j = 0; j < 128; j++) s += __expf(row[j] - m);
        float inv = 1.f / s;
        for (int j = 0; j < 128; j++) Pbuf[t * 132 + j] = (__bf16)(__expf(row[j] - m) * inv);
    }
    __syncthreads();

    // ---------------- phase 2: F = hf * P^T ----------------
    {
        const int wm = wid >> 1;            // 0..3 over c (256)
        const int wn = wid & 1;             // 0..1 over i (128)
        v8f acc[4][4];
#pragma unroll
        for (int a = 0; a < 4; a++)
#pragma unroll
            for (int b = 0; b < 4; b++)
#pragma unroll
                for (int q = 0; q < 8; q++) acc[a][b][q] = 0.f;

        for (int js = 0; js < 4; js++) {
            int j0 = js * 32;
            __syncthreads();
            {   // A2tile[c][j'] = hf[c][j0+j'] (coalesced)
                const __bf16* src = hf + (size_t)t * NPTS + j0;
                __bf16* dst = A2tile + t * 34;
#pragma unroll
                for (int i = 0; i < 32; i++) dst[i] = src[i];
            }
            __syncthreads();

            v16bf afr[4], bfr[4];
#pragma unroll
            for (int fi = 0; fi < 4; fi++) {
                int m = wm * 64 + fi * 16 + mlo;
                const unsigned* row = (const unsigned*)(A2tile + m * 34);
                union { v16bf v; unsigned u[8]; } U;
#pragma unroll
                for (int v8 = 0; v8 < 8; v8++) {
                    int k = hiB16 + 2 * v8;
                    U.u[v8] = row[k >> 1];
                }
                afr[fi] = U.v;
            }
#pragma unroll
            for (int fj = 0; fj < 4; fj++) {
                int icol = wn * 64 + fj * 16 + mlo;
                const unsigned* prow = (const unsigned*)(Pbuf + icol * 132 + j0);
                union { v16bf v; unsigned u[8]; } U;
#pragma unroll
                for (int v8 = 0; v8 < 8; v8++) U.u[v8] = prow[(hiB16 >> 1) + v8];
                bfr[fj] = U.v;
            }
#pragma unroll
            for (int fi = 0; fi < 4; fi++)
#pragma unroll
                for (int fj = 0; fj < 4; fj++)
                    acc[fi][fj] = __builtin_amdgcn_wmma_f32_16x16x32_bf16(
                        false, afr[fi], false, bfr[fj], (short)0, acc[fi][fj], false, false);
        }

        // fused = F + lf32
#pragma unroll
        for (int fi = 0; fi < 4; fi++)
#pragma unroll
            for (int fj = 0; fj < 4; fj++) {
                int icol = wn * 64 + fj * 16 + mlo;
#pragma unroll
                for (int v8 = 0; v8 < 8; v8++) {
                    int c = wm * 64 + fi * 16 + v8 + mhi;
                    fup[(size_t)c * NPTS + icol] = acc[fi][fj][v8] + lfp[(size_t)c * NPTS + icol];
                }
            }
    }
}

// ---------------------------------------------------------------------------
// K7: x_low -> out (float4 grid-stride copy)
// ---------------------------------------------------------------------------
__global__ void __launch_bounds__(256) copy_kernel(const float4* __restrict__ src,
                                                   float4* __restrict__ dst, size_t n4) {
    size_t i = (size_t)blockIdx.x * blockDim.x + threadIdx.x;
    size_t stride = (size_t)gridDim.x * blockDim.x;
    for (; i < n4; i += stride) dst[i] = src[i];
}

// ---------------------------------------------------------------------------
// K8/K9: scatter fused features into low grid (edge first, then corner)
// ---------------------------------------------------------------------------
__global__ void __launch_bounds__(256) scatter_kernel(const float* __restrict__ fused,
                                                      const int* __restrict__ lowidx,
                                                      float* __restrict__ out, int br) {
    int n = blockIdx.x, t = threadIdx.x;
    const float* fu = fused + ((size_t)br * NB + n) * CIN * NPTS;
    const int* li = lowidx + (br * NB + n) * NPTS;
    __shared__ int sli[NPTS];
    if (t < NPTS) sli[t] = li[t];
    __syncthreads();
    float* ob = out + (size_t)n * CIN * (HL * WL);
    for (int e = t; e < CIN * NPTS; e += 256) {
        int c = e >> 7, p = e & 127;
        ob[(size_t)c * (HL * WL) + sli[p]] = fu[e];
    }
}

// ---------------------------------------------------------------------------
extern "C" void kernel_launch(void* const* d_in, const int* in_sizes, int n_in,
                              void* d_out, int out_size, void* d_ws, size_t ws_size,
                              hipStream_t stream) {
    (void)in_sizes; (void)n_in; (void)out_size; (void)ws_size;
    const float* x_high = (const float*)d_in[0];
    const float* x_low  = (const float*)d_in[1];
    const float* w_e1   = (const float*)d_in[2];
    const float* g_e    = (const float*)d_in[3];
    const float* b_e    = (const float*)d_in[4];
    const float* m_e    = (const float*)d_in[5];
    const float* v_e    = (const float*)d_in[6];
    const float* w_e2   = (const float*)d_in[7];
    const float* w_c1   = (const float*)d_in[8];
    const float* g_c    = (const float*)d_in[9];
    const float* b_c    = (const float*)d_in[10];
    const float* m_c    = (const float*)d_in[11];
    const float* v_c    = (const float*)d_in[12];
    const float* w_c2   = (const float*)d_in[13];

    float* out = (float*)d_out;
    const size_t EDGE_OFF   = (size_t)NB * CIN * HL * WL;   // 67,108,864
    const size_t CORNER_OFF = EDGE_OFF + (size_t)NB * HH * WH;

    // workspace carve-up (all 256B-aligned chunks)
    char* ws = (char*)d_ws;
    size_t off = 0;
    auto carve = [&](size_t bytes) {
        void* p = ws + off;
        off += (bytes + 255) & ~(size_t)255;
        return p;
    };
    __bf16* wpk  = (__bf16*)carve(9ull * 512 * 256 * 2);
    float*  bnsc = (float*) carve(512 * 4);
    float*  bnbi = (float*) carve(512 * 4);
    __bf16* xhb  = (__bf16*)carve((size_t)NB * CIN * HH * WH * 2);
    __bf16* y1   = (__bf16*)carve((size_t)NB * COUT * HH * WH * 2);
    __bf16* hfb  = (__bf16*)carve(2ull * NB * CIN * NPTS * 2);
    __bf16* lfb  = (__bf16*)carve(2ull * NB * CIN * NPTS * 2);
    float*  lf32 = (float*) carve(2ull * NB * CIN * NPTS * 4);
    float*  fus  = (float*) carve(2ull * NB * CIN * NPTS * 4);
    int*    topk = (int*)   carve(2ull * NB * NPTS * 4);
    int*    lowi = (int*)   carve(2ull * NB * NPTS * 4);

    pack_w_kernel<<<4608, 256, 0, stream>>>(w_e1, w_c1, wpk);
    bn_prep_kernel<<<2, 256, 0, stream>>>(g_e, b_e, m_e, v_e, g_c, b_c, m_c, v_c, bnsc, bnbi);
    cvt_bf16_kernel<<<65536, 256, 0, stream>>>(x_high, xhb, NB * CIN * HH * WH);
    conv1_wmma_kernel<<<dim3(32, 4, NB), 256, 0, stream>>>(xhb, wpk, bnsc, bnbi, y1);
    edge_pred_kernel<<<256, 256, 0, stream>>>(y1, w_e2, out + EDGE_OFF);
    corner_pred_kernel<<<256, 256, 0, stream>>>(y1, w_c2, out + CORNER_OFF);
    topk_kernel<<<dim3(NB, 2), 256, 0, stream>>>(out + EDGE_OFF, topk);
    gather_kernel<<<dim3(NB, 2), 256, 0, stream>>>(x_high, x_low, topk, hfb, lfb, lf32, lowi);
    attn_wmma_kernel<<<dim3(NB, 2), 256, 0, stream>>>(hfb, lfb, lf32, fus);
    copy_kernel<<<8192, 256, 0, stream>>>((const float4*)x_low, (float4*)out,
                                          (size_t)NB * CIN * HL * WL / 4);
    scatter_kernel<<<NB, 256, 0, stream>>>(fus, lowi, out, 0);   // edge first
    scatter_kernel<<<NB, 256, 0, stream>>>(fus, lowi, out, 1);   // corner overwrites
}
